// NaiveLaplaceKANLayer_67078799229654
// MI455X (gfx1250) — matrix operand order repaired and checked by reading, not compiled, not run on hardware
//
#include <hip/hip_runtime.h>
#include <hip/hip_bf16.h>

// Laplace-KAN layer as an F16-WMMA GEMM:
//   y(B,O) = [exp(-x*lam) | exp(+x*lam)] (B x 2*I*G)  @  W^T (2*I*G x O) + bias
// B=4096, I=256, O=256, G=16  ->  M=4096, N=256, K=8192 (two sign halves)
// Compute-bound (~2150 FLOP/byte): f16 WMMA with f32 accumulation.

#define BDIM  4096
#define IDIM  256
#define ODIM  256
#define KHALF 4096           // I*G per sign half
#define PITCH 40             // LDS row pitch in halves (32 data + 8 pad = 80 B)
#define LOG2E 1.4426950408889634f

typedef __attribute__((ext_vector_type(16))) _Float16 v16h;
typedef __attribute__((ext_vector_type(8)))  _Float16 v8h;
typedef __attribute__((ext_vector_type(4)))  _Float16 v4h;
typedef __attribute__((ext_vector_type(8)))  float    v8f;
typedef __attribute__((ext_vector_type(4)))  float    v4f;

union V16 { v16h v; v8h h[2]; };

// A-matrix fragment (16x32 f16), contiguous K layout (matches frag_b):
// lanes 0-15: M=lane, K 0..15 contiguous
// lanes 16-31: M=lane-16, K 16..31
static __device__ __forceinline__ v16h frag_a(const _Float16* rowbase, int lane) {
  const int r  = lane & 15;
  const int hi = lane >> 4;
  const _Float16* p = rowbase + r * PITCH + hi * 16;
  V16 f;
  f.h[0] = *(const v8h*)(p);        // 16 B: K {0..7} or {16..23}
  f.h[1] = *(const v8h*)(p + 8);    // 16 B: K {8..15} or {24..31}
  return f.v;
}

// B-matrix fragment (32x16 f16): lanes 0-15: N=lane, K 0..15 contiguous;
// lanes 16-31: N=lane-16, K 16..31. (rowbase indexed by N, K-major rows)
static __device__ __forceinline__ v16h frag_b(const _Float16* rowbase, int lane) {
  const int r  = lane & 15;
  const int hi = lane >> 4;
  const _Float16* p = rowbase + r * PITCH + hi * 16;
  V16 f;
  f.h[0] = *(const v8h*)(p);
  f.h[1] = *(const v8h*)(p + 8);
  return f.v;
}

__global__ void __launch_bounds__(256)
laplace_kan_init(const float* __restrict__ bias, float* __restrict__ y) {
  const int idx = blockIdx.x * 256 + threadIdx.x;   // 4096 blocks * 256 = B*O
  y[idx] = bias[idx & (ODIM - 1)];
}

__global__ void __launch_bounds__(256)
laplace_kan_gemm(const float* __restrict__ x,
                 const float* __restrict__ W,     // (2, O, I*G) row-major == B^T
                 float* __restrict__ y) {
  // double-buffered tiles: 2 * (128*80B + 128*80B) = 40 KB LDS
  __shared__ __align__(16) _Float16 sA[2][128 * PITCH];  // 128 M-rows x 32 K halves
  __shared__ __align__(16) _Float16 sB[2][128 * PITCH];  // 128 N-rows x 32 K halves

  const int nblk = blockIdx.x;          // 0..1   (128 cols each)
  const int mblk = blockIdx.y;          // 0..31  (128 rows each)
  const int s    = blockIdx.z;          // 0: exp(-e) w/ W[0], 1: exp(+e) w/ W[1]
  const int t    = threadIdx.x;
  const int lane = t & 31;
  const int wave = t >> 5;              // 0..7
  const int wm   = wave & 1;            // wave M offset = wm*64
  const int wn   = wave >> 1;           // wave N offset = wn*32

  const float sgn  = s ? 1.0f : -1.0f;
  const int mbase  = mblk * 128;
  const int nbase  = nblk * 128;
  const float* Ws  = W + (size_t)(s * ODIM + nbase) * KHALF;

  v8f acc[4][2];
#pragma unroll
  for (int a = 0; a < 4; ++a)
#pragma unroll
    for (int b = 0; b < 2; ++b) acc[a][b] = (v8f)0.0f;

  // staging index split
  const int fm = t >> 1;                // feature row 0..127
  const int fi = t & 1;                 // which of the two i's in this K-step
  const int bn = t >> 3;                // coeff row group 0..31
  const int bk = t & 7;                 // coeff 4-float chunk 0..7

  // ---- staging helper: fill buffer `buf` with tile at K offset `kk` ----
  auto stage = [&](int kk, int buf) {
    const int i0 = kk >> 4;
    // A: f16 Laplace features, exp(sgn*x*lam_g) = exp2(xs * (lam_g*log2e))
    const float xs = sgn * x[(size_t)(mbase + fm) * IDIM + i0 + fi];
    v8h h0, h1;
#pragma unroll
    for (int g = 0; g < 8; ++g)
      h0[g] = (_Float16)__builtin_amdgcn_exp2f(xs * (LOG2E * (0.1f + 0.06f * (float)g)));
#pragma unroll
    for (int g = 0; g < 8; ++g)
      h1[g] = (_Float16)__builtin_amdgcn_exp2f(xs * (LOG2E * (0.1f + 0.06f * (float)(g + 8))));
    _Float16* dstA = &sA[buf][fm * PITCH + fi * 16];
    *(v8h*)(dstA)     = h0;
    *(v8h*)(dstA + 8) = h1;
    // B: convert 128x32 f32 coeff tile to f16
#pragma unroll
    for (int rep = 0; rep < 4; ++rep) {
      const int n = rep * 32 + bn;
      const float* gp = Ws + (size_t)n * KHALF + kk + bk * 4;
      v4f wv = *(const v4f*)gp;
      if (kk + 32 < KHALF) __builtin_prefetch(gp + 32, 0, 1);   // next K-chunk
      v4h hb;
#pragma unroll
      for (int c = 0; c < 4; ++c) hb[c] = (_Float16)wv[c];
      *(v4h*)&sB[buf][n * PITCH + bk * 4] = hb;
    }
  };

  // prologue: stage first tile
  stage(0, 0);
  __syncthreads();

  int cur = 0;
  for (int kk = 0; kk < KHALF; kk += 32) {
    // stage next tile into the alternate buffer (global-load latency hides
    // behind this iteration's WMMAs; one barrier per K-step)
    if (kk + 32 < KHALF) stage(kk + 32, cur ^ 1);

    // per-wave 64x32 tile from current buffer: 12 ds_load_b128 + 8 WMMAs
    v16h bfr0 = frag_b(&sB[cur][(wn * 32 + 0)  * PITCH], lane);
    v16h bfr1 = frag_b(&sB[cur][(wn * 32 + 16) * PITCH], lane);
#pragma unroll
    for (int tm = 0; tm < 4; ++tm) {
      v16h afr = frag_a(&sA[cur][(wm * 64 + tm * 16) * PITCH], lane);
      acc[tm][0] = __builtin_amdgcn_wmma_f32_16x16x32_f16(
          false, afr, false, bfr0, (short)0, acc[tm][0], false, false);
      acc[tm][1] = __builtin_amdgcn_wmma_f32_16x16x32_f16(
          false, afr, false, bfr1, (short)0, acc[tm][1], false, false);
    }

    __syncthreads();
    cur ^= 1;
  }

  // ---- epilogue: C layout (v8f tile: row = v + 8*(lane>=16), col = lane%16) ----
  const int col0 = nbase + wn * 32 + (lane & 15);
  const int rowh = (lane >> 4) * 8;
#pragma unroll
  for (int tm = 0; tm < 4; ++tm)
#pragma unroll
    for (int tn = 0; tn < 2; ++tn)
#pragma unroll
      for (int v = 0; v < 8; ++v) {
        const int row = mbase + wm * 64 + tm * 16 + rowh + v;
        const int col = col0 + tn * 16;
        atomicAdd(&y[(size_t)row * ODIM + col], acc[tm][tn][v]);
      }
}

extern "C" void kernel_launch(void* const* d_in, const int* in_sizes, int n_in,
                              void* d_out, int out_size, void* d_ws, size_t ws_size,
                              hipStream_t stream) {
  const float* x    = (const float*)d_in[0];   // (B, I)      f32
  const float* W    = (const float*)d_in[1];   // (2, O, I, G) f32
  const float* bias = (const float*)d_in[2];   // (1, O)      f32
  float* y = (float*)d_out;                    // (B, O)      f32

  // y = bias (also clears the 0xAA poison), then both sign-halves atomically add.
  laplace_kan_init<<<dim3(BDIM * ODIM / 256), dim3(256), 0, stream>>>(bias, y);

  dim3 grid(ODIM / 128, BDIM / 128, 2);        // (nblk, mblk, sign) = 2 x 32 x 2
  laplace_kan_gemm<<<grid, dim3(256), 0, stream>>>(x, W, y);
}
